// PerceiverAttention_81286551044416
// MI455X (gfx1250) — matrix operand order repaired and checked by reading, not compiled, not run on hardware
//
#include <hip/hip_runtime.h>

// ---------------------------------------------------------------------------
// PerceiverAttention on MI455X (gfx1250): bf16 WMMA pipeline.
// B=4, LQ=512, LKV=4096, IN_DIM=1024, H=16, D=64, INNER=1024.
// ---------------------------------------------------------------------------

#define B_    4
#define LQ_   512
#define LKV_  4096
#define IND_  1024
#define NH_   16
#define DH_   64
#define INNER_ 1024

typedef __attribute__((ext_vector_type(16))) __bf16 v16bf;
typedef __attribute__((ext_vector_type(8)))  float  v8f;
typedef __attribute__((ext_vector_type(4)))  float  f32x4;

#define WMMA_BF16(a, b, c) \
  __builtin_amdgcn_wmma_f32_16x16x32_bf16(false, (a), false, (b), (short)0, (c), false, false)

// A-fragment (16x32 bf16, MxK): lane holds row (lane&15); element e <-> K =
// (e<8 ? e : e+8) + 8*(lane>=16). Two contiguous 8-elem (16B) runs.
static __device__ inline v16bf load_frag_a(const __bf16* rowp, int half) {
  union { v16bf v; f32x4 f[2]; } u;
  u.f[0] = *(const f32x4*)(rowp + half * 8);
  u.f[1] = *(const f32x4*)(rowp + half * 8 + 16);
  return u.v;
}

// B-fragment (32x16 bf16, KxN) loaded from N-major ("B^T", row = N, contiguous
// along K): lane holds col (lane&15); element e <-> K =
// (e<8 ? e : e+8) + 8*(lane>=16).
static __device__ inline v16bf load_frag_b(const __bf16* rowp, int half) {
  union { v16bf v; f32x4 f[2]; } u;
  u.f[0] = *(const f32x4*)(rowp + half * 8);
  u.f[1] = *(const f32x4*)(rowp + half * 8 + 16);
  return u.v;
}

// ---------------------------------------------------------------------------
// Elementwise f32 -> bf16 convert
// ---------------------------------------------------------------------------
__global__ __launch_bounds__(256) void cvt_bf16_kernel(const float* __restrict__ in,
                                                       __bf16* __restrict__ out,
                                                       size_t n) {
  size_t i = (size_t)blockIdx.x * blockDim.x + threadIdx.x;
  if (i < n) out[i] = (__bf16)in[i];
}

// Transpose + convert: in is K x N (f32, row-major), out is N x K (bf16).
__global__ __launch_bounds__(256) void transpose_bf16_kernel(const float* __restrict__ in,
                                                             __bf16* __restrict__ out,
                                                             int K, int N) {
  size_t i = (size_t)blockIdx.x * blockDim.x + threadIdx.x;
  size_t total = (size_t)K * N;
  if (i < total) {
    int k = (int)(i / N);
    int n = (int)(i % N);
    out[(size_t)n * K + k] = (__bf16)in[i];
  }
}

// ---------------------------------------------------------------------------
// Generic WMMA GEMM: C[M,N] = A[M,K] * BT[N,K]^T.
// Block = 8 waves; wave tile 32x128 (2 A-frags x 8 B-frags = 16 WMMA / k-step)
// => block tile 128x256. Two WMMAs consume each B fragment (reg-level reuse).
// ---------------------------------------------------------------------------
template <typename OutT>
__global__ __launch_bounds__(256) void gemm_bf16_nt(const __bf16* __restrict__ A,
                                                    const __bf16* __restrict__ BT,
                                                    OutT* __restrict__ C,
                                                    int M, int N, int K) {
  const int wave = threadIdx.x >> 5;
  const int lane = threadIdx.x & 31;
  const int half = lane >> 4;
  const int rsel = lane & 15;
  const int m0 = (blockIdx.x * 4 + (wave & 3)) * 32;
  const int n0 = (blockIdx.y * 2 + (wave >> 2)) * 128;

  v8f acc0[8] = {};
  v8f acc1[8] = {};
  const __bf16* aptr0 = A + (size_t)(m0 + rsel) * K;
  const __bf16* aptr1 = A + (size_t)(m0 + 16 + rsel) * K;

  for (int k0 = 0; k0 < K; k0 += 32) {
    v16bf a0 = load_frag_a(aptr0 + k0, half);
    v16bf a1 = load_frag_a(aptr1 + k0, half);
#pragma unroll
    for (int j = 0; j < 8; ++j) {
      v16bf b = load_frag_b(BT + (size_t)(n0 + j * 16 + rsel) * K + k0, half);
      acc0[j] = WMMA_BF16(a0, b, acc0[j]);
      acc1[j] = WMMA_BF16(a1, b, acc1[j]);
    }
  }

#pragma unroll
  for (int j = 0; j < 8; ++j) {
    int col = n0 + j * 16 + rsel;
#pragma unroll
    for (int v = 0; v < 8; ++v) {
      int row = m0 + v + 8 * half;
      C[(size_t)row * N + col] = (OutT)acc0[j][v];
      C[(size_t)(row + 16) * N + col] = (OutT)acc1[j][v];
    }
  }
}

// ---------------------------------------------------------------------------
// KV projection GEMM: kvp = kv * Wkv, epilogue scatters into per-head
//   kbuf[b,h,k,d]   (row-major in d -> QK^T B-fragments)
//   vbuf[b,h,d,k]   (row-major in k -> PV   B-fragments, i.e. V^T)
// M = B*LKV = 16384, N = 2*INNER = 2048, K = 1024. Wave tile 32x128.
// ---------------------------------------------------------------------------
__global__ __launch_bounds__(256) void gemm_kv_proj(const __bf16* __restrict__ A,
                                                    const __bf16* __restrict__ BT,
                                                    __bf16* __restrict__ kbuf,
                                                    __bf16* __restrict__ vbuf) {
  const int K = IND_;
  const int wave = threadIdx.x >> 5;
  const int lane = threadIdx.x & 31;
  const int half = lane >> 4;
  const int rsel = lane & 15;
  const int m0 = (blockIdx.x * 4 + (wave & 3)) * 32;
  const int n0 = (blockIdx.y * 2 + (wave >> 2)) * 128;

  v8f acc0[8] = {};
  v8f acc1[8] = {};
  const __bf16* aptr0 = A + (size_t)(m0 + rsel) * K;
  const __bf16* aptr1 = A + (size_t)(m0 + 16 + rsel) * K;

  for (int k0 = 0; k0 < K; k0 += 32) {
    v16bf a0 = load_frag_a(aptr0 + k0, half);
    v16bf a1 = load_frag_a(aptr1 + k0, half);
#pragma unroll
    for (int j = 0; j < 8; ++j) {
      v16bf b = load_frag_b(BT + (size_t)(n0 + j * 16 + rsel) * K + k0, half);
      acc0[j] = WMMA_BF16(a0, b, acc0[j]);
      acc1[j] = WMMA_BF16(a1, b, acc1[j]);
    }
  }

#pragma unroll
  for (int j = 0; j < 8; ++j) {
    int n = n0 + j * 16 + rsel;           // global output column in [0,2048)
    int h = n >> 7;                        // head
    int w = n & 127;                       // within head: [0,64)=K, [64,128)=V
    int d = w & 63;
    bool isV = (w >= 64);
#pragma unroll
    for (int sub = 0; sub < 2; ++sub) {
#pragma unroll
      for (int v = 0; v < 8; ++v) {
        int r = m0 + sub * 16 + v + 8 * half;  // global row = b*LKV + kidx
        int bb = r >> 12;
        int kidx = r & (LKV_ - 1);
        __bf16 val = (__bf16)(sub ? acc1[j][v] : acc0[j][v]);
        if (isV) {
          vbuf[(((size_t)(bb * NH_ + h)) * DH_ + d) * LKV_ + kidx] = val;
        } else {
          kbuf[(((size_t)(bb * NH_ + h)) * LKV_ + kidx) * DH_ + d] = val;
        }
      }
    }
  }
}

// ---------------------------------------------------------------------------
// Flash attention. Block = 8 waves; each wave owns one 16-row Q tile of one
// (b, h) and streams LKV in 32-key chunks with online softmax.
// cbuf[b*LQ + q, h*64 + d]  (bf16 context, feeds final GEMM).
// ---------------------------------------------------------------------------
__global__ __launch_bounds__(256) void flash_attn_kernel(const __bf16* __restrict__ qp,
                                                         const __bf16* __restrict__ kbuf,
                                                         const __bf16* __restrict__ vbuf,
                                                         const unsigned char* __restrict__ mask,
                                                         __bf16* __restrict__ cbuf) {
  __shared__ __attribute__((aligned(16))) __bf16 plds[8][16 * 32];

  const int wave = threadIdx.x >> 5;
  const int lane = threadIdx.x & 31;
  const int half = lane >> 4;
  const int rsel = lane & 15;

  const int job = blockIdx.x * 8 + wave;  // 2048 jobs = B*H*(LQ/16)
  const int qt = job & 31;
  const int h = (job >> 5) & (NH_ - 1);
  const int b = job >> 9;
  const int q0 = qt * 16;

  // Q A-fragments (16 rows x 64 d = 2 K-chunks), loaded once.
  const __bf16* qbase = qp + ((size_t)(b * LQ_) + q0 + rsel) * INNER_ + h * DH_;
  v16bf qf0 = load_frag_a(qbase, half);
  v16bf qf1 = load_frag_a(qbase + 32, half);

  const __bf16* Kb = kbuf + ((size_t)(b * NH_ + h)) * LKV_ * DH_;
  const __bf16* Vb = vbuf + ((size_t)(b * NH_ + h)) * DH_ * LKV_;
  const unsigned char* mk = mask + (size_t)b * LKV_;

  float m_r[8], l_r[8];
  v8f o[4] = {};
#pragma unroll
  for (int v = 0; v < 8; ++v) { m_r[v] = -1e30f; l_r[v] = 0.0f; }

  for (int kc = 0; kc < LKV_; kc += 32) {
    // gfx1250 prefetch of the next 32-key chunk (K rows + V^T rows) so the
    // L2->L0 latency overlaps the softmax VALU phase below.
    if (kc + 32 < LKV_) {
      const __bf16* pk = Kb + (size_t)(kc + 32 + lane) * DH_;
      __builtin_prefetch(pk, 0, 3);
      __builtin_prefetch(pk + 32, 0, 3);
      __builtin_prefetch(Vb + (size_t)lane * LKV_ + kc + 32, 0, 3);
      __builtin_prefetch(Vb + (size_t)(lane + 32) * LKV_ + kc + 32, 0, 3);
    }

    // ---- S = Q K^T for two 16-key tiles (contraction over d = 64) ----
    v8f s0 = {}, s1 = {};
    {
      v16bf kb = load_frag_b(Kb + (size_t)(kc + rsel) * DH_, half);
      s0 = WMMA_BF16(qf0, kb, s0);
      kb = load_frag_b(Kb + (size_t)(kc + rsel) * DH_ + 32, half);
      s0 = WMMA_BF16(qf1, kb, s0);
      kb = load_frag_b(Kb + (size_t)(kc + 16 + rsel) * DH_, half);
      s1 = WMMA_BF16(qf0, kb, s1);
      kb = load_frag_b(Kb + (size_t)(kc + 16 + rsel) * DH_ + 32, half);
      s1 = WMMA_BF16(qf1, kb, s1);
    }

    const bool mz0 = mk[kc + rsel] != 0;        // lane owns key column kc+rsel
    const bool mz1 = mk[kc + 16 + rsel] != 0;

    // ---- online softmax per row (row v+8*half lives in this lane's half) ----
#pragma unroll
    for (int v = 0; v < 8; ++v) {
      float a0 = mz0 ? -1e30f : s0[v] * 0.125f;
      float a1 = mz1 ? -1e30f : s1[v] * 0.125f;

      float mx = fmaxf(a0, a1);
#pragma unroll
      for (int d = 1; d < 16; d <<= 1) mx = fmaxf(mx, __shfl_xor(mx, d, 32));
      float mn = fmaxf(m_r[v], mx);
      float fac = __expf(m_r[v] - mn);
      float p0 = __expf(a0 - mn);
      float p1 = __expf(a1 - mn);
      float rs = p0 + p1;
#pragma unroll
      for (int d = 1; d < 16; d <<= 1) rs += __shfl_xor(rs, d, 32);
      l_r[v] = l_r[v] * fac + rs;
      m_r[v] = mn;
#pragma unroll
      for (int j = 0; j < 4; ++j) o[j][v] *= fac;

      // stage P (C-layout -> LDS 16x32 tile, row-major)
      plds[wave][(v + 8 * half) * 32 + rsel] = (__bf16)p0;
      plds[wave][(v + 8 * half) * 32 + 16 + rsel] = (__bf16)p1;
    }

    // intra-wave LDS store->load ordering (CDNA5 split DS counter)
    asm volatile("s_wait_dscnt 0" ::: "memory");

    // reload P as A-fragment (16x32 over this key chunk)
    v16bf pf = load_frag_a((const __bf16*)&plds[wave][rsel * 32], half);

    // ---- O += P V (contraction over 32 keys, 4 d-tiles) ----
#pragma unroll
    for (int j = 0; j < 4; ++j) {
      v16bf vf = load_frag_b(Vb + (size_t)(j * 16 + rsel) * LKV_ + kc, half);
      o[j] = WMMA_BF16(pf, vf, o[j]);
    }
  }

  // ---- normalize and emit context (bf16, [B*LQ, INNER]) ----
  float inv[8];
#pragma unroll
  for (int v = 0; v < 8; ++v) inv[v] = 1.0f / fmaxf(l_r[v], 1e-20f);
#pragma unroll
  for (int j = 0; j < 4; ++j) {
#pragma unroll
    for (int v = 0; v < 8; ++v) {
      int row = q0 + v + 8 * half;
      int col = h * DH_ + j * 16 + rsel;
      cbuf[((size_t)(b * LQ_) + row) * INNER_ + col] = (__bf16)(o[j][v] * inv[v]);
    }
  }
}

// ---------------------------------------------------------------------------
// Host-side launch
// ---------------------------------------------------------------------------
extern "C" void kernel_launch(void* const* d_in, const int* in_sizes, int n_in,
                              void* d_out, int out_size, void* d_ws, size_t ws_size,
                              hipStream_t stream) {
  const float* q = (const float*)d_in[0];    // [B, LQ, IND]
  const float* kv = (const float*)d_in[1];   // [B, LKV, IND]
  const unsigned char* mask = (const unsigned char*)d_in[2];  // [B,1,1,LKV] bool
  const float* Wq = (const float*)d_in[3];   // [IND, INNER]
  const float* Wkv = (const float*)d_in[4];  // [IND, 2*INNER]
  const float* Wo = (const float*)d_in[5];   // [INNER, IND]
  float* out = (float*)d_out;                // [B, LQ, IND]

  // workspace carving (bf16 buffers), 256B aligned
  size_t off = 0;
  auto carve = [&](size_t bytes) {
    void* p = (char*)d_ws + off;
    off += (bytes + 255) & ~(size_t)255;
    return p;
  };
  const size_t nQ = (size_t)B_ * LQ_ * IND_;       // 2M
  const size_t nKV = (size_t)B_ * LKV_ * IND_;     // 16M
  __bf16* qbf  = (__bf16*)carve(nQ * 2);
  __bf16* kvbf = (__bf16*)carve(nKV * 2);
  __bf16* WqT  = (__bf16*)carve((size_t)IND_ * INNER_ * 2);
  __bf16* WkvT = (__bf16*)carve((size_t)IND_ * 2 * INNER_ * 2);
  __bf16* WoT  = (__bf16*)carve((size_t)INNER_ * IND_ * 2);
  __bf16* qproj = (__bf16*)carve((size_t)B_ * LQ_ * INNER_ * 2);
  __bf16* kbuf = (__bf16*)carve((size_t)B_ * NH_ * LKV_ * DH_ * 2);
  __bf16* vbuf = (__bf16*)carve((size_t)B_ * NH_ * DH_ * LKV_ * 2);
  __bf16* cbuf = (__bf16*)carve((size_t)B_ * LQ_ * INNER_ * 2);

  // 1) converts / weight transposes
  cvt_bf16_kernel<<<(unsigned)((nQ + 255) / 256), 256, 0, stream>>>(q, qbf, nQ);
  cvt_bf16_kernel<<<(unsigned)((nKV + 255) / 256), 256, 0, stream>>>(kv, kvbf, nKV);
  transpose_bf16_kernel<<<(unsigned)(((size_t)IND_ * INNER_ + 255) / 256), 256, 0, stream>>>(
      Wq, WqT, IND_, INNER_);
  transpose_bf16_kernel<<<(unsigned)(((size_t)IND_ * 2 * INNER_ + 255) / 256), 256, 0, stream>>>(
      Wkv, WkvT, IND_, 2 * INNER_);
  transpose_bf16_kernel<<<(unsigned)(((size_t)INNER_ * IND_ + 255) / 256), 256, 0, stream>>>(
      Wo, WoT, INNER_, IND_);

  // 2) q projection: [2048,1024] = [2048,1024] x [1024,1024]^T-stored
  gemm_bf16_nt<__bf16><<<dim3((B_ * LQ_) / 128, INNER_ / 256), 256, 0, stream>>>(
      qbf, WqT, qproj, B_ * LQ_, INNER_, IND_);

  // 3) kv projection with per-head K / V^T scatter: M=16384, N=2048
  gemm_kv_proj<<<dim3((B_ * LKV_) / 128, (2 * INNER_) / 256), 256, 0, stream>>>(
      kvbf, WkvT, kbuf, vbuf);

  // 4) flash attention: 2048 wave-jobs, 8 per block
  flash_attn_kernel<<<(B_ * NH_ * (LQ_ / 16)) / 8, 256, 0, stream>>>(
      qproj, kbuf, vbuf, mask, cbuf);

  // 5) output projection to f32: [2048,1024] = [2048,1024] x [1024,1024]^T-stored
  gemm_bf16_nt<float><<<dim3((B_ * LQ_) / 128, IND_ / 256), 256, 0, stream>>>(
      cbuf, WoT, out, B_ * LQ_, IND_, INNER_);
}
